// KPConv_38225208934610
// MI455X (gfx1250) — matrix operand order repaired and checked by reading, not compiled, not run on hardware
//
#include <hip/hip_runtime.h>
#include <hip/hip_bf16.h>

// ---------------------------------------------------------------------------
// KPConv on MI455X (gfx1250): two fused WMMA GEMM stages per 16-query tile.
//   Stage A (per query):  WF(16x64)  = nw(16x32,f16) @ nf(32x64,f16)   [pad K=15->16]
//   Stage B (per tile):   out(16x64) = WF(16x960,f16) @ W(960x64,f16)
// Both via v_wmma_f32_16x16x32_f16 (f32 accumulate). s_feats (12.8MB) and
// weights (240KB) are L2-resident (192MB L2), so HBM traffic is streaming-only
// and the kernel is matrix-op bound. Weights are pre-swizzled to the per-lane
// B-fragment layout in d_ws by a tiny prep kernel so the phase-3 inner loop is
// 2x global_load_b128 + 2x ds_load_b128 + v_wmma.
// ---------------------------------------------------------------------------

typedef __attribute__((ext_vector_type(16))) _Float16 v16h;
typedef __attribute__((ext_vector_type(8)))  _Float16 v8h;
typedef __attribute__((ext_vector_type(8)))  float    v8f;

union Frag {
    v16h v;
    v8h  h[2];
    _Float16 e[16];
};

#define KP_H      32     // neighbors
#define KP_C      64     // input channels
#define KP_CO     64     // output channels
#define KP_K      15     // kernel points
#define KP_TILE_M 16     // queries per workgroup
#define KP_INV_SIGMA (1.0f / 0.7f)

// LDS layout (dynamic shared):
//  nwA  [16][16][32] f16  : influence weights, A-matrix order (rows = kernel k)
//  nfT  [16][64][32] f16  : gathered features transposed [m][c][h]
//  WFA  [16][960]    f16  : stage-A output, A-matrix rows for stage B
//  kp   [45]         f32  : kernel points
#define OFF_NWA  0
#define OFF_NFT  16384          // 16*16*32*2
#define OFF_WFA  81920          // OFF_NFT + 16*64*32*2
#define OFF_KP   112640         // OFF_WFA + 16*960*2
#define SMEM_BYTES 112832       // OFF_KP + 192

// Pre-swizzled weights in d_ws: 30 K-chunks x 4 N-tiles = 120 fragments,
// each 32 lanes x 32 bytes (16 f16) = 1KB  ->  120 KB total.
#define W_SWZ_BYTES (120u * 32u * 32u)

static __device__ __forceinline__ unsigned pack2(float a, float b) {
    union { _Float16 h[2]; unsigned u; } p;
    p.h[0] = (_Float16)a;
    p.h[1] = (_Float16)b;
    return p.u;
}

// --- prep: convert weights (960x64 f32 row-major) to f16 fragment layout ----
__global__ __launch_bounds__(32)
void KPConv_prep_weights(const float* __restrict__ W, unsigned* __restrict__ ws)
{
    const int frag = blockIdx.x;        // kk*4 + nt, 120 fragments
    const int kk   = frag >> 2;
    const int nt   = frag & 3;
    const int lane = threadIdx.x;
    const int col  = lane & 15;
    const int kB   = (lane < 16) ? 0 : 16;   // B: lanes0-15 K0..15, lanes16-31 K16..31
    const int n    = nt * 16 + col;

    unsigned vals[8];
    #pragma unroll
    for (int j = 0; j < 8; ++j) {
        const int pr = kk * 32 + kB + 2 * j;
        vals[j] = pack2(W[(size_t)pr * KP_CO + n], W[(size_t)(pr + 1) * KP_CO + n]);
    }
    uint4* dst = (uint4*)(ws + ((size_t)frag * 32 + lane) * 8);
    uint4 u0; u0.x = vals[0]; u0.y = vals[1]; u0.z = vals[2]; u0.w = vals[3];
    uint4 u1; u1.x = vals[4]; u1.y = vals[5]; u1.z = vals[6]; u1.w = vals[7];
    dst[0] = u0;
    dst[1] = u1;
}

template <bool USE_WS>
__global__ __launch_bounds__(128)
void KPConv_gfx1250_kernel(const float* __restrict__ q_pts,
                           const float* __restrict__ s_pts,
                           const float* __restrict__ s_feats,
                           const int*   __restrict__ neighb_inds,
                           const float* __restrict__ kernel_points,
                           const float* __restrict__ weights,
                           const char*  __restrict__ wswz,
                           float* __restrict__ out,
                           int M, int N)
{
    extern __shared__ char smem[];
    unsigned* nwA32 = (unsigned*)(smem + OFF_NWA);   // packed h-pairs
    unsigned* nfT32 = (unsigned*)(smem + OFF_NFT);   // packed h-pairs
    _Float16* nwA   = (_Float16*)(smem + OFF_NWA);
    _Float16* nfT   = (_Float16*)(smem + OFF_NFT);
    _Float16* WFA   = (_Float16*)(smem + OFF_WFA);
    float*    kp    = (float*)   (smem + OFF_KP);

    const int tid  = threadIdx.x;
    const int lane = tid & 31;
    const int w    = tid >> 5;          // wave id 0..3
    const int m0   = blockIdx.x * KP_TILE_M;

    if (tid < KP_K * 3) kp[tid] = kernel_points[tid];
    __syncthreads();

    // ------- Phase 1: influence weights + gathered features (packed h-pairs)
    for (int p = tid; p < KP_TILE_M * (KP_H / 2); p += 128) {
        const int mi = p >> 4;          // query in tile
        const int h2 = p & 15;          // neighbor pair index (h = 2*h2, 2*h2+1)
        const int gm = m0 + mi;

        int nb0 = -1, nb1 = -1;
        if (gm < M) {
            nb0 = neighb_inds[(size_t)gm * KP_H + 2 * h2 + 0];
            nb1 = neighb_inds[(size_t)gm * KP_H + 2 * h2 + 1];
        }
        const bool v0 = (nb0 >= 0) && (nb0 < N);
        const bool v1 = (nb1 >= 0) && (nb1 < N);

        float qx = 0.f, qy = 0.f, qz = 0.f;
        if (gm < M) {
            qx = q_pts[(size_t)gm * 3 + 0];
            qy = q_pts[(size_t)gm * 3 + 1];
            qz = q_pts[(size_t)gm * 3 + 2];
        }
        float n0x = 0.f, n0y = 0.f, n0z = 0.f, n1x = 0.f, n1y = 0.f, n1z = 0.f;
        if (v0) {
            n0x = s_pts[(size_t)nb0 * 3 + 0] - qx;
            n0y = s_pts[(size_t)nb0 * 3 + 1] - qy;
            n0z = s_pts[(size_t)nb0 * 3 + 2] - qz;
        }
        if (v1) {
            n1x = s_pts[(size_t)nb1 * 3 + 0] - qx;
            n1y = s_pts[(size_t)nb1 * 3 + 1] - qy;
            n1z = s_pts[(size_t)nb1 * 3 + 2] - qz;
        }

        // nw rows (kernel idx k), packed pair store; row k=15 zero padding.
        #pragma unroll
        for (int k = 0; k < 16; ++k) {
            float a = 0.f, b = 0.f;
            if (k < KP_K) {
                const float kx = kp[k * 3 + 0], ky = kp[k * 3 + 1], kz = kp[k * 3 + 2];
                if (v0) {
                    const float dx = n0x - kx, dy = n0y - ky, dz = n0z - kz;
                    a = fmaxf(0.f, 1.f - __builtin_sqrtf(dx*dx + dy*dy + dz*dz) * KP_INV_SIGMA);
                }
                if (v1) {
                    const float dx = n1x - kx, dy = n1y - ky, dz = n1z - kz;
                    b = fmaxf(0.f, 1.f - __builtin_sqrtf(dx*dx + dy*dy + dz*dz) * KP_INV_SIGMA);
                }
            }
            nwA32[(mi * 16 + k) * (KP_H / 2) + h2] = pack2(a, b);
        }

        // transposed feature gather: two neighbor rows, packed b32 stores.
        const float4* s0 = (const float4*)(s_feats + (size_t)(v0 ? nb0 : 0) * KP_C);
        const float4* s1 = (const float4*)(s_feats + (size_t)(v1 ? nb1 : 0) * KP_C);
        #pragma unroll
        for (int c4 = 0; c4 < KP_C / 4; ++c4) {
            float4 f0 = s0[c4];
            float4 f1 = s1[c4];
            if (!v0) { f0.x = f0.y = f0.z = f0.w = 0.f; }
            if (!v1) { f1.x = f1.y = f1.z = f1.w = 0.f; }
            const int c = c4 * 4;
            nfT32[(mi * KP_C + c + 0) * (KP_H / 2) + h2] = pack2(f0.x, f1.x);
            nfT32[(mi * KP_C + c + 1) * (KP_H / 2) + h2] = pack2(f0.y, f1.y);
            nfT32[(mi * KP_C + c + 2) * (KP_H / 2) + h2] = pack2(f0.z, f1.z);
            nfT32[(mi * KP_C + c + 3) * (KP_H / 2) + h2] = pack2(f0.w, f1.w);
        }
    }
    __syncthreads();

    // ------- Phase 2: WF = nw @ nf  (per query, 4 N-tiles) ------------------
    {
        const int col = lane & 15;
        const int hbA = (lane < 16) ? 0 : 16;   // A: lanes0-15 K0..15, lanes16-31 K16..31
        const int hbB = (lane < 16) ? 0 : 16;   // B: lanes0-15 K0..15, lanes16-31 K16..31
        for (int mi = w * 4; mi < w * 4 + 4; ++mi) {
            Frag a;
            const _Float16* ar = nwA + (mi * 16 + col) * KP_H;
            a.h[0] = *(const v8h*)(ar + hbA);
            a.h[1] = *(const v8h*)(ar + hbA + 8);
            #pragma unroll
            for (int nt = 0; nt < 4; ++nt) {
                Frag b;
                const _Float16* br = nfT + (mi * KP_C + nt * 16 + col) * KP_H;
                b.h[0] = *(const v8h*)(br + hbB);
                b.h[1] = *(const v8h*)(br + hbB + 8);
                v8f z = {};
                v8f d = __builtin_amdgcn_wmma_f32_16x16x32_f16(
                    false, a.v, false, b.v, (short)0, z, false, false);
                // D rows = kernel idx: vgpr r -> k=r (lanes0-15) / k=r+8 (lanes16-31)
                #pragma unroll
                for (int r = 0; r < 8; ++r) {
                    const int k = (lane < 16) ? r : r + 8;
                    if (k < KP_K)
                        WFA[mi * (KP_K * KP_C) + k * KP_C + nt * 16 + col] =
                            (_Float16)d[r];
                }
            }
        }
    }
    __syncthreads();

    // ------- Phase 3: out = WF(16x960) @ W(960x64) --------------------------
    {
        const int col = lane & 15;
        const int n   = w * 16 + col;           // output channel (wave owns N-tile w)
        const int hbA = (lane < 16) ? 0 : 16;
        const int kB  = (lane < 16) ? 0 : 16;
        v8f acc = {};
        #pragma unroll 2
        for (int kk = 0; kk < (KP_K * KP_C) / 32; ++kk) {   // 30 K-chunks
            Frag a;
            const _Float16* ar = WFA + col * (KP_K * KP_C) + kk * 32;
            a.h[0] = *(const v8h*)(ar + hbA);
            a.h[1] = *(const v8h*)(ar + hbA + 8);
            Frag b;
            if (USE_WS) {
                // pre-swizzled f16 fragments: 2x b128 per lane
                const v8h* wp = (const v8h*)(wswz +
                    ((size_t)(kk * 4 + w) * 32 + lane) * 32);
                b.h[0] = wp[0];
                b.h[1] = wp[1];
            } else {
                #pragma unroll
                for (int j = 0; j < 8; ++j) {
                    const int pr = kk * 32 + kB + 2 * j;
                    b.e[2*j+0] = (_Float16)weights[(size_t)pr * KP_CO + n];
                    b.e[2*j+1] = (_Float16)weights[(size_t)(pr + 1) * KP_CO + n];
                }
            }
            acc = __builtin_amdgcn_wmma_f32_16x16x32_f16(
                false, a.v, false, b.v, (short)0, acc, false, false);
        }
        #pragma unroll
        for (int r = 0; r < 8; ++r) {
            const int m  = (lane < 16) ? r : r + 8;
            const int gm = m0 + m;
            if (gm < M)
                out[(size_t)gm * KP_CO + n] = acc[r];
        }
    }
}

extern "C" void kernel_launch(void* const* d_in, const int* in_sizes, int n_in,
                              void* d_out, int out_size, void* d_ws, size_t ws_size,
                              hipStream_t stream) {
    const float* q_pts   = (const float*)d_in[0];
    const float* s_pts   = (const float*)d_in[1];
    const float* s_feats = (const float*)d_in[2];
    const int*   inds    = (const int*)  d_in[3];
    const float* kpts    = (const float*)d_in[4];
    const float* wts     = (const float*)d_in[5];
    float*       out     = (float*)d_out;

    const int M = in_sizes[0] / 3;
    const int N = in_sizes[1] / 3;

    hipFuncSetAttribute((const void*)KPConv_gfx1250_kernel<true>,
                        hipFuncAttributeMaxDynamicSharedMemorySize, SMEM_BYTES);
    hipFuncSetAttribute((const void*)KPConv_gfx1250_kernel<false>,
                        hipFuncAttributeMaxDynamicSharedMemorySize, SMEM_BYTES);

    const int grid = (M + KP_TILE_M - 1) / KP_TILE_M;
    if (ws_size >= (size_t)W_SWZ_BYTES) {
        KPConv_prep_weights<<<120, 32, 0, stream>>>(wts, (unsigned*)d_ws);
        KPConv_gfx1250_kernel<true><<<grid, 128, SMEM_BYTES, stream>>>(
            q_pts, s_pts, s_feats, inds, kpts, wts, (const char*)d_ws, out, M, N);
    } else {
        KPConv_gfx1250_kernel<false><<<grid, 128, SMEM_BYTES, stream>>>(
            q_pts, s_pts, s_feats, inds, kpts, wts, (const char*)d_ws, out, M, N);
    }
}
